// LSTMForecaster_72078141161923
// MI455X (gfx1250) — matrix operand order repaired and checked by reading, not compiled, not run on hardware
//
#include <hip/hip_runtime.h>
#include <hip/hip_bf16.h>

// ---------------------------------------------------------------------------
// Fused 2-layer LSTM (H=64) + FC head for MI455X (gfx1250, wave32, WMMA).
//   gates = x_t @ W_ih^T + h_{t-1} @ W_hh^T + b   via v_wmma_f32_16x16x32_bf16
//   c,h updated in fp32; h stored bf16 in LDS (double buffered).
// Block = 128 threads (4 waves) handles a 16-row batch tile for all 512 steps.
// Wave w owns h-columns [16w,16w+16) and gate tiles {w, 4+w, 8+w, 12+w}.
// All weight B-fragments are register-resident for the whole time loop.
// Biases live in persistent accumulator-layout f32x8 vectors and enter the
// gate GEMMs as the C operand of the first WMMA (zero per-step init cost).
// Activations are branchless native v_exp2/v_rcp sequences.
// ---------------------------------------------------------------------------

#define BATCH   2048
#define SEQ_T   512
#define HDIM    64
#define BT      16      // batch rows per block (WMMA M)
#define NWAVES  4
#define ROWU    33      // padded row stride in uints (64 bf16 + 2 pad)

typedef __attribute__((ext_vector_type(16))) __bf16 bf16x16;
typedef __attribute__((ext_vector_type(8)))  float  f32x8;

union Frag { bf16x16 v; unsigned int u[8]; };

__device__ __forceinline__ unsigned int f2bf(float f) {
    unsigned int x = __float_as_uint(f);
    return (x + 0x7FFFu + ((x >> 16) & 1u)) >> 16;   // round-to-nearest-even
}
__device__ __forceinline__ unsigned int pack2(float lo, float hi) {
    return f2bf(lo) | (f2bf(hi) << 16);
}
__device__ __forceinline__ float bf2f(unsigned int b16) {
    return __uint_as_float(b16 << 16);
}

// Branchless native activations: v_mul + v_exp_f32 + v_add + v_rcp_f32 (+fma).
// Limits are handled by IEEE inf/zero propagation (no range branches).
#define LOG2E  1.4426950408889634f
__device__ __forceinline__ float sigm_fast(float x) {
    return __builtin_amdgcn_rcpf(1.0f + __builtin_amdgcn_exp2f(-LOG2E * x));
}
__device__ __forceinline__ float tanh_fast(float x) {
    // tanh(x) = 1 - 2/(exp(2x)+1); exp2(+inf)->inf, rcp(inf)->0 => +/-1 limits
    return __builtin_fmaf(
        -2.0f,
        __builtin_amdgcn_rcpf(1.0f + __builtin_amdgcn_exp2f((2.0f * LOG2E) * x)),
        1.0f);
}

__device__ __forceinline__ f32x8 wmma_bf16(const Frag& a, const Frag& b, f32x8 c) {
    return __builtin_amdgcn_wmma_f32_16x16x32_bf16(
        /*neg_a=*/false, a.v, /*neg_b=*/false, b.v,
        /*c_mod=*/(short)0, c, /*reuse_a=*/false, /*reuse_b=*/false);
}

// B-fragment (32x16, K x N) for gate-tile with column base nb, K range [k0,k0+32).
// W is row-major [256][K] (gates @ W^T => B[k][n] = W[nb+n][k]).
// Layout: n = lane&15; lanes 0-15 hold K k0..k0+15, lanes 16-31 hold k0+16..k0+31.
__device__ __forceinline__ Frag load_wfrag(const float* __restrict__ W, int K,
                                           int nb, int k0, int lane) {
    Frag f;
    const int n  = nb + (lane & 15);
    const int kb = k0 + (((lane >> 4) & 1) ? 16 : 0);
    const float* row = W + (size_t)n * K;
#pragma unroll
    for (int j = 0; j < 8; ++j) {
        int k = kb + 2 * j;
        float lo = (k     < K) ? row[k]     : 0.0f;
        float hi = (k + 1 < K) ? row[k + 1] : 0.0f;
        f.u[j] = pack2(lo, hi);
    }
    return f;
}

// A-fragment (16x32, M x K) from LDS bf16 rows packed as uints (pair k,k+1).
// Layout: m = lane&15; lanes 0-15 kb=0, lanes 16-31 kb=16;
// VGPR j: k = kb + 2*j, relative to k0.
__device__ __forceinline__ Frag load_afrag(const unsigned int* sh, int k0, int lane) {
    Frag f;
    const int m  = lane & 15;
    const int kb = k0 + (((lane >> 4) & 1) ? 16 : 0);
    const unsigned int* row = sh + m * ROWU;
#pragma unroll
    for (int j = 0; j < 8; ++j) {
        int k = kb + 2 * j;
        f.u[j] = row[k >> 1];
    }
    return f;
}

// Nonlinear cell update for one wave's 16x16 tile; writes bf16 h into LDS.
// Biases are pre-folded into acc. Element r -> (m = mofs + r, n = lane&15).
__device__ __forceinline__ void cell_update(const f32x8 acc[4],
                                            f32x8& c, unsigned short* hsh,
                                            int hcol, int mofs) {
#pragma unroll
    for (int r = 0; r < 8; ++r) {
        float iv = sigm_fast(acc[0][r]);
        float fv = sigm_fast(acc[1][r]);
        float gv = tanh_fast(acc[2][r]);
        float ov = sigm_fast(acc[3][r]);
        float cn = fv * c[r] + iv * gv;
        c[r] = cn;
        hsh[(mofs + r) * (2 * ROWU) + hcol] = (unsigned short)f2bf(ov * tanh_fast(cn));
    }
}

__global__ __launch_bounds__(BT * NWAVES * 2)
void lstm2_fused_kernel(const float* __restrict__ x,
                        const float* __restrict__ W_ih0,
                        const float* __restrict__ W_hh0,
                        const float* __restrict__ b0,
                        const float* __restrict__ W_ih1,
                        const float* __restrict__ W_hh1,
                        const float* __restrict__ b1,
                        const float* __restrict__ W_fc,
                        const float* __restrict__ b_fc,
                        float* __restrict__ out) {
    __shared__ unsigned int h0_sh[2][BT * ROWU];
    __shared__ unsigned int h1_sh[2][BT * ROWU];

    const int tid  = threadIdx.x;
    const int lane = tid & 31;
    const int wave = tid >> 5;          // 0..3
    const int b0r  = blockIdx.x * BT;   // batch row base

    // Zero both h buffers.
    for (int i = tid; i < BT * ROWU; i += BT * NWAVES * 2) {
        h0_sh[0][i] = 0u; h0_sh[1][i] = 0u;
        h1_sh[0][i] = 0u; h1_sh[1][i] = 0u;
    }

    // ---- Register-resident weight fragments (loaded once) ----
    const int nb[4] = { 16 * wave, 64 + 16 * wave, 128 + 16 * wave, 192 + 16 * wave };

    Frag Wih0f[4];            // K=4 padded into a single K=32 fragment
    Frag Whh0f[4][2];         // K=64 -> 2 halves
    Frag Wih1f[4][2];
    Frag Whh1f[4][2];
#pragma unroll
    for (int g = 0; g < 4; ++g) {
        Wih0f[g]    = load_wfrag(W_ih0, 4,  nb[g], 0,  lane);
        Whh0f[g][0] = load_wfrag(W_hh0, 64, nb[g], 0,  lane);
        Whh0f[g][1] = load_wfrag(W_hh0, 64, nb[g], 32, lane);
        Wih1f[g][0] = load_wfrag(W_ih1, 64, nb[g], 0,  lane);
        Wih1f[g][1] = load_wfrag(W_ih1, 64, nb[g], 32, lane);
        Whh1f[g][0] = load_wfrag(W_hh1, 64, nb[g], 0,  lane);
        Whh1f[g][1] = load_wfrag(W_hh1, 64, nb[g], 32, lane);
    }

    // Persistent bias accumulators (C-operand of the first WMMA per gate).
    // C layout: every element of a lane shares the same column n = lane&15.
    const int col = lane & 15;
    f32x8 biasv0[4], biasv1[4];
#pragma unroll
    for (int g = 0; g < 4; ++g) {
        float v0 = b0[nb[g] + col];
        float v1 = b1[nb[g] + col];
#pragma unroll
        for (int r = 0; r < 8; ++r) { biasv0[g][r] = v0; biasv1[g][r] = v1; }
    }

    f32x8 c0; f32x8 c1;
#pragma unroll
    for (int r = 0; r < 8; ++r) { c0[r] = 0.0f; c1[r] = 0.0f; }

    const int mofs = ((lane >> 4) & 1) ? 8 : 0;
    const int hcol = 16 * wave + col;
    const float* xrow = x + ((size_t)(b0r + col)) * SEQ_T * 4;  // all lanes converged

    __syncthreads();

    int p = 0;  // read buffer parity
    for (int t = 0; t < SEQ_T; ++t) {
        // ---- x A-fragment: only k<4 (lanes 0-15, VGPRs 0-1) are nonzero ----
        const float4 xv = *reinterpret_cast<const float4*>(xrow + 4 * t);
        int tp = t + 16; tp = (tp < SEQ_T) ? tp : (SEQ_T - 1);   // branchless clamp
        __builtin_prefetch(xrow + 4 * tp, 0, 0);
        Frag xA;
        xA.u[0] = (lane < 16) ? pack2(xv.x, xv.y) : 0u;
        xA.u[1] = (lane < 16) ? pack2(xv.z, xv.w) : 0u;
#pragma unroll
        for (int j = 2; j < 8; ++j) xA.u[j] = 0u;

        // ---- Layer 0: gates = x@Wih0^T + h0_prev@Whh0^T + b0 ----
        Frag h0A0 = load_afrag(h0_sh[p], 0,  lane);
        Frag h0A1 = load_afrag(h0_sh[p], 32, lane);

        f32x8 acc[4];
#pragma unroll
        for (int g = 0; g < 4; ++g) {
            acc[g] = wmma_bf16(xA,   Wih0f[g],    biasv0[g]);   // C = bias
            acc[g] = wmma_bf16(h0A0, Whh0f[g][0], acc[g]);
            acc[g] = wmma_bf16(h0A1, Whh0f[g][1], acc[g]);
        }
        cell_update(acc, c0, (unsigned short*)h0_sh[1 - p], hcol, mofs);
        __syncthreads();   // h0_new visible to all waves

        // ---- Layer 1: gates = h0_new@Wih1^T + h1_prev@Whh1^T + b1 ----
        Frag h0nA0 = load_afrag(h0_sh[1 - p], 0,  lane);
        Frag h0nA1 = load_afrag(h0_sh[1 - p], 32, lane);
        Frag h1A0  = load_afrag(h1_sh[p],     0,  lane);
        Frag h1A1  = load_afrag(h1_sh[p],     32, lane);
#pragma unroll
        for (int g = 0; g < 4; ++g) {
            acc[g] = wmma_bf16(h0nA0, Wih1f[g][0], biasv1[g]);  // C = bias
            acc[g] = wmma_bf16(h0nA1, Wih1f[g][1], acc[g]);
            acc[g] = wmma_bf16(h1A0,  Whh1f[g][0], acc[g]);
            acc[g] = wmma_bf16(h1A1,  Whh1f[g][1], acc[g]);
        }
        cell_update(acc, c1, (unsigned short*)h1_sh[1 - p], hcol, mofs);
        __syncthreads();   // h1_new visible; all reads of buffer p retired

        p ^= 1;
    }

    // ---- FC head: out[b] = h1_last[b] . W_fc + b_fc ----
    if (tid < BT) {
        const unsigned short* hrow = (const unsigned short*)h1_sh[p] + tid * (2 * ROWU);
        float s = b_fc[0];
#pragma unroll
        for (int k = 0; k < HDIM; ++k) s += bf2f(hrow[k]) * W_fc[k];
        out[b0r + tid] = s;
    }
}

extern "C" void kernel_launch(void* const* d_in, const int* in_sizes, int n_in,
                              void* d_out, int out_size, void* d_ws, size_t ws_size,
                              hipStream_t stream) {
    (void)in_sizes; (void)n_in; (void)d_ws; (void)ws_size; (void)out_size;
    const float* x     = (const float*)d_in[0];
    const float* W_ih0 = (const float*)d_in[1];
    const float* W_hh0 = (const float*)d_in[2];
    const float* b0    = (const float*)d_in[3];
    const float* W_ih1 = (const float*)d_in[4];
    const float* W_hh1 = (const float*)d_in[5];
    const float* b1    = (const float*)d_in[6];
    const float* W_fc  = (const float*)d_in[7];
    const float* b_fc  = (const float*)d_in[8];
    float* out = (float*)d_out;

    dim3 grid(BATCH / BT);          // 128 blocks
    dim3 block(BT * NWAVES * 2);    // 128 threads = 4 waves (wave32)
    lstm2_fused_kernel<<<grid, block, 0, stream>>>(
        x, W_ih0, W_hh0, b0, W_ih1, W_hh1, b1, W_fc, b_fc, out);
}
